// LSTM_36086315221096
// MI455X (gfx1250) — matrix operand order repaired and checked by reading, not compiled, not run on hardware
//
#include <hip/hip_runtime.h>
#include <hip/hip_bf16.h>
#include <cstdint>

// ---- CDNA5 WMMA types ----
typedef __attribute__((ext_vector_type(16))) __bf16 v16bf;
typedef __attribute__((ext_vector_type(8)))  float  v8f;

union FragBF {
  uint4 q[2];   // two 16-byte LDS loads
  v16bf v;      // 16 bf16 = 32 bytes per lane
};

// f32 pair -> packed bf16 in one v_perm_b32 (+ two rounding adds).
// Round-to-nearest (ties away from zero): add 0x8000 then truncate.
__device__ __forceinline__ uint32_t pk_bf16(float a, float b) {
  union { float f; uint32_t u; } xa, xb;
  xa.f = a; xb.f = b;
  const uint32_t ua = xa.u + 0x8000u;
  const uint32_t ub = xb.u + 0x8000u;
#if __has_builtin(__builtin_amdgcn_perm)
  // {hi=ub, lo=ua}; selector bytes (LSB first): ua[2], ua[3], ub[2], ub[3]
  return __builtin_amdgcn_perm(ub, ua, 0x07060302u);
#else
  return (ua >> 16) | (ub & 0xFFFF0000u);
#endif
}

__device__ __forceinline__ float sigmoid_f(float z) {
  return 1.0f / (1.0f + __expf(-z));
}
__device__ __forceinline__ float tanh_f(float z) {
  float e = __expf(-2.0f * fabsf(z));   // in (0,1], numerically stable
  float t = (1.0f - e) / (1.0f + e);
  return copysignf(t, z);
}

// Problem dims (fixed by the reference)
constexpr int Bsz = 4096;          // batch
constexpr int Hn  = 2048;          // hidden size
constexpr int Kn  = 4096;          // H + I reduction dim
constexpr int TILE_M = 128;        // batch rows per workgroup
constexpr int TILE_N = 64;         // hidden cols per workgroup (x4 gates)
constexpr int TILE_K = 32;         // K per WMMA step
constexpr int NKT    = Kn / TILE_K;

__global__ __launch_bounds__(256) void lstm_wmma_kernel(
    const float* __restrict__ x,  const float* __restrict__ hidden,
    const float* __restrict__ ct,
    const float* __restrict__ Wf, const float* __restrict__ bf,
    const float* __restrict__ Wi, const float* __restrict__ bi,
    const float* __restrict__ Wo, const float* __restrict__ bo,
    const float* __restrict__ Wg, const float* __restrict__ bg,
    float* __restrict__ out)
{
  // double-buffered bf16 tiles (packed u32 pairs); 16 u32 per 32-K row
  __shared__ uint32_t sA[2][TILE_M * (TILE_K / 2)];      // 2 x 8 KB
  __shared__ uint32_t sB[2][4][TILE_N * (TILE_K / 2)];   // 2 x 16 KB

  const int tid  = threadIdx.x;
  const int lane = tid & 31;
  const int wv   = tid >> 5;        // wave 0..7
  const int wm   = wv >> 1;         // 0..3 : 32-row stripe
  const int wn   = wv & 1;          // 0..1 : 32-col stripe
  const int half = lane >> 4;       // which K-half this lane holds
  const int l16  = lane & 15;

  const int m0 = blockIdx.x * TILE_M;   // batch-row base
  const int h0 = blockIdx.y * TILE_N;   // hidden-col base

  const float* Wp[4] = { Wf, Wi, Wo, Wg };

  // staging assignment for this thread (fixed across K-steps)
  const int arow = tid >> 3;            // A: rows tid/8 + {0,32,64,96}
  const int ac4  = tid & 7;             // A/B: float4 column
  const int brow = tid >> 3;            // B: rows tid/8 + {0,32}

  v8f acc[4][2][2];
  #pragma unroll
  for (int g = 0; g < 4; ++g)
    #pragma unroll
    for (int mi = 0; mi < 2; ++mi)
      #pragma unroll
      for (int ni = 0; ni < 2; ++ni)
        acc[g][mi][ni] = v8f{};

  float4 rA[4];
  float4 rB[4][2];

  // ---- global load of K-tile kt into registers ----
  auto load_tile = [&](int kt) {
    const int k0 = kt * TILE_K;
    const float* Abase = (k0 < Hn) ? (hidden + k0) : (x + (k0 - Hn));
    #pragma unroll
    for (int p = 0; p < 4; ++p) {
      rA[p] = *reinterpret_cast<const float4*>(
          Abase + (size_t)(m0 + arow + p * 32) * 2048 + ac4 * 4);
    }
    #pragma unroll
    for (int g = 0; g < 4; ++g) {
      const float* Wb = Wp[g] + (size_t)h0 * Kn + k0;
      #pragma unroll
      for (int p = 0; p < 2; ++p) {
        rB[g][p] = *reinterpret_cast<const float4*>(
            Wb + (size_t)(brow + p * 32) * Kn + ac4 * 4);
      }
    }
  };

  // ---- convert staged registers and store into LDS buffer `cur` ----
  auto store_tile = [&](int cur) {
    #pragma unroll
    for (int p = 0; p < 4; ++p) {
      const int row = arow + p * 32;
      sA[cur][row * 16 + ac4 * 2]     = pk_bf16(rA[p].x, rA[p].y);
      sA[cur][row * 16 + ac4 * 2 + 1] = pk_bf16(rA[p].z, rA[p].w);
    }
    #pragma unroll
    for (int g = 0; g < 4; ++g) {
      #pragma unroll
      for (int p = 0; p < 2; ++p) {
        const int n = brow + p * 32;
        sB[cur][g][n * 16 + ac4 * 2]     = pk_bf16(rB[g][p].x, rB[g][p].y);
        sB[cur][g][n * 16 + ac4 * 2 + 1] = pk_bf16(rB[g][p].z, rB[g][p].w);
      }
    }
  };

  load_tile(0);        // prologue: stage K-tile 0
  int cur = 0;

  for (int kt = 0; kt < NKT; ++kt) {
    store_tile(cur);           // regs (tile kt) -> LDS[cur]

    if (kt + 1 < NKT)          // stage tile kt+1; loads fly under the WMMAs
      load_tile(kt + 1);

    // prefetch tile kt+2 into near caches, two stages ahead
    if (kt + 2 < NKT) {
      const int k2 = (kt + 2) * TILE_K;
      const float* An = (k2 < Hn) ? (hidden + k2) : (x + (k2 - Hn));
      __builtin_prefetch(An + (size_t)(m0 + arow) * 2048 + ac4 * 4, 0, 3);
      __builtin_prefetch(Wp[tid & 3] + (size_t)(h0 + brow) * Kn + k2, 0, 3);
    }

    __syncthreads();           // LDS[cur] visible; prior reads of it retired

    // ---- A fragments (16x32 bf16, ISA Sec 7.12.2 layout):
    // lanes 0-15 hold K {0..7,16..23}, lanes 16-31 hold K {8..15,24..31}
    FragBF a[2];
    #pragma unroll
    for (int mi = 0; mi < 2; ++mi) {
      const uint4* pa = reinterpret_cast<const uint4*>(
          &sA[cur][(wm * 32 + mi * 16 + l16) * 16]);
      a[mi].q[0] = pa[half];
      a[mi].q[1] = pa[half + 2];
    }

    // ---- per gate: B fragments (32x16: lane holds W row = B column;
  // lanes 0-15 K {0..7,16..23}, lanes 16-31 K {8..15,24..31}), then WMMAs
    #pragma unroll
    for (int g = 0; g < 4; ++g) {
      #pragma unroll
      for (int ni = 0; ni < 2; ++ni) {
        FragBF b;
        const uint4* pb = reinterpret_cast<const uint4*>(
            &sB[cur][g][(wn * 32 + ni * 16 + l16) * 16]);
        b.q[0] = pb[half];
        b.q[1] = pb[half + 2];
        #pragma unroll
        for (int mi = 0; mi < 2; ++mi) {
          acc[g][mi][ni] = __builtin_amdgcn_wmma_f32_16x16x32_bf16(
              false, a[mi].v, false, b.v, (short)0, acc[g][mi][ni],
              false, false);
        }
      }
    }

    cur ^= 1;
  }

  // ---- fused LSTM epilogue ----
  // C/D layout: VGPR r -> M = r + 8*(lane>=16); N = lane%16
  float* outH = out;
  float* outC = out + (size_t)Bsz * Hn;

  #pragma unroll
  for (int ni = 0; ni < 2; ++ni) {
    const int col = h0 + wn * 32 + ni * 16 + l16;
    const float vbf = bf[col], vbi = bi[col], vbo = bo[col], vbg = bg[col];
    #pragma unroll
    for (int mi = 0; mi < 2; ++mi) {
      #pragma unroll
      for (int r = 0; r < 8; ++r) {
        const int row = m0 + wm * 32 + mi * 16 + half * 8 + r;
        const float fz = sigmoid_f(acc[0][mi][ni][r] + vbf);
        const float iz = sigmoid_f(acc[1][mi][ni][r] + vbi);
        const float oz = sigmoid_f(acc[2][mi][ni][r] + vbo);
        const float gz = tanh_f   (acc[3][mi][ni][r] + vbg);
        const float cprev = ct[(size_t)row * Hn + col];
        const float cn = fz * cprev + iz * gz;
        const float hn = oz * tanh_f(cn);
        outH[(size_t)row * Hn + col] = hn;
        outC[(size_t)row * Hn + col] = cn;
      }
    }
  }
}

extern "C" void kernel_launch(void* const* d_in, const int* in_sizes, int n_in,
                              void* d_out, int out_size, void* d_ws, size_t ws_size,
                              hipStream_t stream) {
  const float* x      = (const float*)d_in[0];
  const float* hidden = (const float*)d_in[1];
  const float* ct     = (const float*)d_in[2];
  const float* Wf     = (const float*)d_in[3];
  const float* bfv    = (const float*)d_in[4];
  const float* Wi     = (const float*)d_in[5];
  const float* biv    = (const float*)d_in[6];
  const float* Wo     = (const float*)d_in[7];
  const float* bov    = (const float*)d_in[8];
  const float* Wg     = (const float*)d_in[9];
  const float* bgv    = (const float*)d_in[10];
  float* out = (float*)d_out;

  dim3 grid(Bsz / TILE_M, Hn / TILE_N);   // 32 x 32 = 1024 workgroups
  dim3 block(256);                        // 8 wave32 waves
  lstm_wmma_kernel<<<grid, block, 0, stream>>>(
      x, hidden, ct, Wf, bfv, Wi, biv, Wo, bov, Wg, bgv, out);
}
